// ActorNetwork_69106023793166
// MI455X (gfx1250) — matrix operand order repaired and checked by reading, not compiled, not run on hardware
//
#include <hip/hip_runtime.h>
#include <cstdint>
#include <cstddef>

// Problem constants (from reference)
#define B_  256
#define S_  512
#define I_  256
#define H_  128
#define A_  18
#define G_  384   // 3*H

typedef __attribute__((ext_vector_type(16))) __bf16 v16bf;
typedef __attribute__((ext_vector_type(8)))  float  v8f;

union BF16x16 { uint32_t u[8]; uint4 q[2]; v16bf v; };

// Pack two fp32 -> one dword of bf16 (round-to-nearest-even), lo in [15:0].
__device__ inline uint32_t pack2bf(float lo, float hi) {
  uint32_t a = __float_as_uint(lo);
  uint32_t b = __float_as_uint(hi);
  a = (a + 0x7FFFu + ((a >> 16) & 1u)) >> 16;
  b = (b + 0x7FFFu + ((b >> 16) & 1u)) >> 16;
  return (a & 0xFFFFu) | (b << 16);
}

__device__ inline v8f splat8(float x) { v8f v = {x, x, x, x, x, x, x, x}; return v; }

// Async copy 16 B global -> LDS (CDNA5, tracked by ASYNCcnt).
// dsaddr = low 32 bits of the generic LDS address (flat aperture rule).
__device__ inline void async_g2l_b128(uint32_t lds_byte, const void* gaddr) {
  asm volatile("global_load_async_to_lds_b128 %0, %1, off"
               :: "v"(lds_byte), "v"((uint64_t)(uintptr_t)gaddr)
               : "memory");
}

// ---------------------------------------------------------------------------
// Kernel 1: W_ih fp32 -> bf16 (row-major, K contiguous) into workspace
// ---------------------------------------------------------------------------
__global__ __launch_bounds__(256) void wih_to_bf16(const float* __restrict__ w,
                                                   uint32_t* __restrict__ o) {
  int p = blockIdx.x * 256 + threadIdx.x;           // pair index along K
  if (p < (G_ * I_) / 2) {
    float2 f = ((const float2*)w)[p];
    o[p] = pack2bf(f.x, f.y);
  }
}

// ---------------------------------------------------------------------------
// Kernel 2: x_proj[b,s,:] = obs[b,s,:] @ W_ih^T + b_ih   (bf16 WMMA, f32 acc)
// Grid: (B*S)/128 blocks of 256 threads; each wave owns one 16-row M-tile.
// ---------------------------------------------------------------------------
__global__ __launch_bounds__(256) void xproj_wmma(const float* __restrict__ obs,
                                                  const uint32_t* __restrict__ wih,
                                                  const float* __restrict__ b_ih,
                                                  float* __restrict__ xp) {
  const int lane = threadIdx.x & 31;
  const int wave = threadIdx.x >> 5;
  const int half = lane >> 4;
  const int lm   = lane & 15;
  const size_t rowBase = (size_t)blockIdx.x * 128 + (size_t)wave * 16;

  // Preload A (16x256 bf16) for this wave's M-tile: 8 K-tiles, ISA layout:
  // lane lm holds row m=lm, K = kt*32 + half*8 .. +7 (dw0-3) and +16..+23 (dw4-7)
  const float* arow = obs + (rowBase + lm) * I_;
  BF16x16 a[8];
#pragma unroll
  for (int kt = 0; kt < 8; ++kt) {
    int k0 = kt * 32 + half * 8;
    float4 f0 = *(const float4*)(arow + k0);
    float4 f1 = *(const float4*)(arow + k0 + 4);
    float4 f2 = *(const float4*)(arow + k0 + 16);
    float4 f3 = *(const float4*)(arow + k0 + 20);
    a[kt].u[0] = pack2bf(f0.x, f0.y); a[kt].u[1] = pack2bf(f0.z, f0.w);
    a[kt].u[2] = pack2bf(f1.x, f1.y); a[kt].u[3] = pack2bf(f1.z, f1.w);
    a[kt].u[4] = pack2bf(f2.x, f2.y); a[kt].u[5] = pack2bf(f2.z, f2.w);
    a[kt].u[6] = pack2bf(f3.x, f3.y); a[kt].u[7] = pack2bf(f3.z, f3.w);
  }

#pragma unroll 1
  for (int nt = 0; nt < 24; ++nt) {
    int n = nt * 16 + lm;                            // B column / weight row
    v8f acc = splat8(b_ih[n]);
#pragma unroll
    for (int kt = 0; kt < 8; ++kt) {
      // B tile: lane reads bf16 of weight row n (K = kt*32+half*8 ..+7 and +16..+23)
      int kd = kt * 16 + half * 4;                   // dword offset in row
      const uint4* bp = (const uint4*)(wih + (size_t)n * (I_ / 2) + kd);
      BF16x16 bt; bt.q[0] = bp[0]; bt.q[1] = bp[2];
      acc = __builtin_amdgcn_wmma_f32_16x16x32_bf16(false, a[kt].v, false, bt.v,
                                                    (short)0, acc, false, false);
    }
    // D element (m = half*8+j, n = nt*16+lm)
#pragma unroll
    for (int j = 0; j < 8; ++j) {
      size_t r = rowBase + (size_t)(half * 8 + j);
      xp[r * G_ + n] = acc[j];
    }
  }
}

// ---------------------------------------------------------------------------
// Kernel 3: fused GRU recurrence + head. One WG per 16 batch rows.
// LDS layout (dynamic, 200960 B total):
// ---------------------------------------------------------------------------
#define WHH_DW   68            // dwords per LDS weight/h row (136 bf16, 272 B)
#define SM_WHH   0             // 384 x 68 dw  = 104448 B (W_hh bf16)
#define SM_WHEAD 104448        //  32 x 68 dw  =   8704 B (W_head bf16, rows>=18 zero)
#define SM_HBF   113152        //  16 x 68 dw  =   4352 B (h, bf16 mirror)
#define SM_HF    117504        //  16 x 128 f32=   8192 B (h, fp32)
#define SM_GATES 125696        //  16 x 392 f32=  25088 B (gate pre-activations)
#define SM_XBUF  150784        //  2 x 16 x 392 f32 = 50176 B (x_proj double buffer)
#define XBUF_B   25088         // bytes per x buffer
#define SM_TOTAL 200960
#define GST      392

__device__ inline void load_a_tile(const uint32_t* hbf, int lm, int half, int kt,
                                   BF16x16& a) {
  int kd = kt * 16 + half * 4;                       // elems kt*32+half*8
  const uint32_t* base = hbf + lm * WHH_DW + kd;
  a.q[0] = *(const uint4*)(base);
  a.q[1] = *(const uint4*)(base + 8);                // elems +16..+23
}

__device__ inline void load_b_tile(const uint32_t* wmat, int row, int half, int kt,
                                   BF16x16& b) {
  int kd = kt * 16 + half * 4;                       // elems kt*32+half*8
  const uint32_t* base = wmat + row * WHH_DW + kd;
  b.q[0] = *(const uint4*)(base);
  b.q[1] = *(const uint4*)(base + 8);
}

__device__ inline void head_emit(const uint32_t* hbf, const uint32_t* whead,
                                 float* __restrict__ logits, int bg0, int s_out,
                                 int lane, int ha, float hb) {
  const int half = lane >> 4, lm = lane & 15;
  v8f acc = splat8(hb);
#pragma unroll
  for (int kt = 0; kt < 4; ++kt) {
    BF16x16 a, b;
    load_a_tile(hbf, lm, half, kt, a);
    load_b_tile(whead, ha, half, kt, b);             // rows >= 18 are zeros
    acc = __builtin_amdgcn_wmma_f32_16x16x32_bf16(false, a.v, false, b.v,
                                                  (short)0, acc, false, false);
  }
  if (ha < A_) {
#pragma unroll
    for (int j = 0; j < 8; ++j) {
      size_t r = (size_t)(bg0 + half * 8 + j);
      logits[(r * S_ + (size_t)s_out) * A_ + ha] = acc[j];
    }
  }
}

__global__ __launch_bounds__(256) void gru_head_wmma(
    const float* __restrict__ xp, const float* __restrict__ h0,
    const float* __restrict__ W_hh, const float* __restrict__ b_hh,
    const float* __restrict__ W_head, const float* __restrict__ b_head,
    float* __restrict__ out) {
  extern __shared__ __align__(16) char smem[];
  uint32_t* whh   = (uint32_t*)(smem + SM_WHH);
  uint32_t* whead = (uint32_t*)(smem + SM_WHEAD);
  uint32_t* hbf   = (uint32_t*)(smem + SM_HBF);
  float*    hf    = (float*)(smem + SM_HF);
  float*    gates = (float*)(smem + SM_GATES);

  const int tid  = threadIdx.x;
  const int lane = tid & 31, wave = tid >> 5;
  const int half = lane >> 4, lm = lane & 15;
  const int bg0  = blockIdx.x * 16;

  // ---- stage weights into LDS (fp32 -> bf16) ----
  for (int p = tid; p < (G_ * H_) / 2; p += 256) {   // W_hh: 24576 pairs
    int n = p >> 6, kd = p & 63;
    float2 f = ((const float2*)W_hh)[p];
    whh[n * WHH_DW + kd] = pack2bf(f.x, f.y);
  }
  for (int i = tid; i < 32 * WHH_DW; i += 256) whead[i] = 0u;
  __syncthreads();
  for (int p = tid; p < (A_ * H_) / 2; p += 256) {   // W_head: 1152 pairs
    int n = p >> 6, kd = p & 63;
    float2 f = ((const float2*)W_head)[p];
    whead[n * WHH_DW + kd] = pack2bf(f.x, f.y);
  }
  // ---- init h from h0 ----
  const float* h0p = h0 + (size_t)bg0 * H_;
  for (int p = tid; p < (16 * H_) / 2; p += 256) {
    int m = p >> 6, kd = p & 63;
    float2 f = ((const float2*)h0p)[p];
    hf[m * H_ + kd * 2]     = f.x;
    hf[m * H_ + kd * 2 + 1] = f.y;
    hbf[m * WHH_DW + kd]    = pack2bf(f.x, f.y);
  }
  __syncthreads();

  // per-wave gate biases (3 N-tiles of the 384-wide gate GEMM per wave)
  const int nt0 = wave * 3;
  float gb[3];
#pragma unroll
  for (int i = 0; i < 3; ++i) gb[i] = b_hh[(nt0 + i) * 16 + lm];

  const bool  do_head = (wave < 2);
  const int   ha      = wave * 16 + lm;              // logit column / W_head row
  const float hb      = (do_head && ha < A_) ? b_head[ha] : 0.0f;

  float* logits = out;                               // (B,S,A)
  float* h_n    = out + (size_t)B_ * S_ * A_;        // (B,H)

  const int me = tid >> 4;                           // elementwise row 0..15
  const int c0 = (tid & 15) * 8;                     // elementwise col base
  const float* xrow = xp + (size_t)(bg0 + me) * S_ * G_;

  // ---- prologue: async-stage x for step 0 into buffer 0 ----
  {
    float* xb0 = (float*)(smem + SM_XBUF) + me * GST;
#pragma unroll
    for (int seg = 0; seg < 3; ++seg) {
      const float* g = xrow + seg * H_ + c0;         // s = 0
      float* l = xb0 + seg * H_ + c0;
      async_g2l_b128((uint32_t)(uintptr_t)l,       g);
      async_g2l_b128((uint32_t)(uintptr_t)(l + 4), g + 4);
    }
  }

  for (int s = 0; s < S_; ++s) {
    // ---- Phase W: gate pre-activations g = h @ W_hh^T + b_hh (WMMA) ----
    BF16x16 at[4];
#pragma unroll
    for (int kt = 0; kt < 4; ++kt) load_a_tile(hbf, lm, half, kt, at[kt]);

#pragma unroll
    for (int i = 0; i < 3; ++i) {
      const int row = (nt0 + i) * 16 + lm;
      v8f acc = splat8(gb[i]);
#pragma unroll
      for (int kt = 0; kt < 4; ++kt) {
        BF16x16 bt;
        load_b_tile(whh, row, half, kt, bt);
        acc = __builtin_amdgcn_wmma_f32_16x16x32_bf16(false, at[kt].v, false, bt.v,
                                                      (short)0, acc, false, false);
      }
#pragma unroll
      for (int j = 0; j < 8; ++j)
        gates[(half * 8 + j) * GST + (nt0 + i) * 16 + lm] = acc[j];
    }

    // Head for position s-1 reuses resident h_s (= out[s-1]) as A-matrix.
    if (do_head && s > 0)
      head_emit(hbf, whead, logits, bg0, s - 1, lane, ha, hb);

    __syncthreads();

    // ---- Phase E: nonlinearity + state update (each (m,c) owned by 1 thread) --
    // Issue async stage of x for step s+1 into the other buffer, then wait for
    // this step's batch (in-order completion => asynccnt<=6 means step-s done).
    if (s + 1 < S_) {
      float* xn = (float*)(smem + SM_XBUF + ((s + 1) & 1) * XBUF_B) + me * GST;
      const float* gx = xrow + (size_t)(s + 1) * G_;
#pragma unroll
      for (int seg = 0; seg < 3; ++seg) {
        const float* g = gx + seg * H_ + c0;
        float* l = xn + seg * H_ + c0;
        async_g2l_b128((uint32_t)(uintptr_t)l,       g);
        async_g2l_b128((uint32_t)(uintptr_t)(l + 4), g + 4);
      }
      asm volatile("s_wait_asynccnt 0x6" ::: "memory");
    } else {
      asm volatile("s_wait_asynccnt 0x0" ::: "memory");
    }

    const float* xs = (const float*)(smem + SM_XBUF + (s & 1) * XBUF_B) + me * GST;
    float hv[8];
#pragma unroll
    for (int q = 0; q < 8; ++q) {
      const int c = c0 + q;
      const float pr = xs[c]          + gates[me * GST + c];
      const float pz = xs[H_ + c]     + gates[me * GST + H_ + c];
      const float gn = gates[me * GST + 2 * H_ + c];
      const float r  = 1.0f / (1.0f + __expf(-pr));
      const float z  = 1.0f / (1.0f + __expf(-pz));
      const float nn = tanhf(xs[2 * H_ + c] + r * gn);
      hv[q] = (1.0f - z) * nn + z * hf[me * H_ + c];
    }
#pragma unroll
    for (int q = 0; q < 8; q += 2) {
      hf[me * H_ + c0 + q]     = hv[q];
      hf[me * H_ + c0 + q + 1] = hv[q + 1];
      hbf[me * WHH_DW + (c0 + q) / 2] = pack2bf(hv[q], hv[q + 1]);
    }
    __syncthreads();
  }

  // ---- tail: logits for s = S-1, and h_n = h_S ----
  if (do_head)
    head_emit(hbf, whead, logits, bg0, S_ - 1, lane, ha, hb);
#pragma unroll
  for (int q = 0; q < 8; ++q)
    h_n[(size_t)(bg0 + me) * H_ + c0 + q] = hf[me * H_ + c0 + q];
}

// ---------------------------------------------------------------------------
// Host launcher
// ---------------------------------------------------------------------------
extern "C" void kernel_launch(void* const* d_in, const int* in_sizes, int n_in,
                              void* d_out, int out_size, void* d_ws, size_t ws_size,
                              hipStream_t stream) {
  (void)in_sizes; (void)n_in; (void)out_size; (void)ws_size;
  const float* obs    = (const float*)d_in[0];
  const float* h0     = (const float*)d_in[1];
  const float* W_ih   = (const float*)d_in[2];
  const float* W_hh   = (const float*)d_in[3];
  const float* b_ih   = (const float*)d_in[4];
  const float* b_hh   = (const float*)d_in[5];
  const float* W_head = (const float*)d_in[6];
  const float* b_head = (const float*)d_in[7];
  float* out = (float*)d_out;

  char* ws = (char*)d_ws;
  float*    xproj  = (float*)ws;                                 // B*S*3H fp32
  uint32_t* wih_bf = (uint32_t*)(ws + (size_t)B_ * S_ * G_ * 4); // 3H*I bf16

  wih_to_bf16<<<(G_ * I_ / 2 + 255) / 256, 256, 0, stream>>>(W_ih, wih_bf);
  xproj_wmma<<<(B_ * S_) / 128, 256, 0, stream>>>(obs, wih_bf, b_ih, xproj);
  gru_head_wmma<<<B_ / 16, 256, SM_TOTAL, stream>>>(xproj, h0, W_hh, b_hh,
                                                    W_head, b_head, out);
}
